// SubpixelRefinement_5652176961637
// MI455X (gfx1250) — matrix operand order repaired and checked by reading, not compiled, not run on hardware
//
#include <hip/hip_runtime.h>
#include <hip/hip_bf16.h>

// ---------------------------------------------------------------------------
// ZNCC template matching on MI455X (gfx1250), WMMA-based.
//
//   corr[b,y,x] = (sum_{dy,dx} tnorm[b,dy,dx]*img[b,y+dy,x+dx]) / win_std
//   then per-batch online softmax-argmax + max.
//
// Per 16x16 output tile: split K over the 17 template rows. Each row is a
// Toeplitz matmul  D[m,n] += sum_j A[m,j] * T_dy[j,n],  A = 16x32 image slab,
// T_dy = 32x16 banded matrix from template row dy -> V_WMMA_F32_16X16X32_F16.
// Window sums s1/s2 reuse A (and A*A via v_pk_mul_f16) against a ones-Toeplitz.
// ---------------------------------------------------------------------------

typedef __attribute__((ext_vector_type(16))) _Float16 v16h;
typedef __attribute__((ext_vector_type(8)))  float    v8f;

#define HT 17
#define WI 384
#define LH 368               // 384-17+1
#define TILES 23             // 368/16
#define NTILES (TILES*TILES) // 529
#define TROW 48              // padded template row: [0..15]=0, [16..32]=t, [33..47]=0
#define LDSW 40              // f16 image tile row stride (halfs), 80B -> 16B aligned
#define INV_T 25.0f          // 1/TEMPERATURE
#define INV_N (1.0f/289.0f)

// ---------------- kernel 1: per-batch template normalization ----------------
__global__ void __launch_bounds__(32)
tmpl_norm_kernel(const float* __restrict__ piece, _Float16* __restrict__ tbuf) {
    const int b    = blockIdx.x;
    const int lane = threadIdx.x;
    const float* p = piece + (size_t)b * (HT * HT);

    float s = 0.f, s2 = 0.f;
    for (int i = lane; i < HT * HT; i += 32) {
        float v = p[i];
        s += v; s2 += v * v;
    }
    #pragma unroll
    for (int off = 16; off > 0; off >>= 1) {
        s  += __shfl_xor(s,  off, 32);
        s2 += __shfl_xor(s2, off, 32);
    }
    const float mean = s * INV_N;
    const float var  = (s2 - 289.0f * mean * mean) * (1.0f / 288.0f);  // ddof=1
    float stdv = sqrtf(fmaxf(var, 0.0f));
    stdv = fmaxf(stdv, 1e-6f);
    const float inv = 1.0f / stdv;

    _Float16* out = tbuf + (size_t)b * (HT * TROW);
    for (int idx = lane; idx < HT * TROW; idx += 32) {
        const int row = idx / TROW, c = idx % TROW;
        float v = 0.0f;
        if (c >= 16 && c < 33) v = (p[row * HT + (c - 16)] - mean) * inv;
        out[idx] = (_Float16)v;
    }
}

// ---------------- kernel 2: correlation + online softmax-argmax -------------
__global__ void __launch_bounds__(256, 1)
zncc_kernel(const float* __restrict__ bg, const _Float16* __restrict__ tbuf,
            float* __restrict__ out) {
    __shared__ __align__(16) _Float16 sh_t[HT * TROW];
    __shared__ __align__(16) _Float16 sh_img[8][32 * LDSW];  // one 32x32 f16 tile per wave
    __shared__ float sh_red[256 * 4];

    const int b    = blockIdx.x;
    const int tid  = threadIdx.x;
    const int wave = tid >> 5;
    const int lane = tid & 31;

    // stage normalized, zero-padded template rows once per block
    for (int i = tid; i < HT * TROW; i += 256)
        sh_t[i] = tbuf[(size_t)b * (HT * TROW) + i];
    __syncthreads();

    const int n     = lane & 15;   // output column within tile (B/C/D lane mapping)
    const int hi    = lane >> 4;
    const int kbase = hi * 16;     // B-fragment K range: lanes<16 -> K 0..15, else 16..31

    // B fragments (Toeplitz of template rows): contiguous 16-half window
    v16h barr[HT];
    #pragma unroll
    for (int dy = 0; dy < HT; ++dy) {
        const int off = dy * TROW + 16 + kbase - n;   // in [1, 32]
        v16h t;
        #pragma unroll
        for (int i = 0; i < 16; ++i) t[i] = sh_t[off + i];
        barr[dy] = t;
    }
    // ones-Toeplitz fragment for box sums
    v16h bones;
    #pragma unroll
    for (int i = 0; i < 16; ++i) {
        const int k = kbase + i;
        bones[i] = (k >= n && (k - n) <= 16) ? (_Float16)1.0f : (_Float16)0.0f;
    }

    const float* img = bg + (size_t)b * (WI * WI);
    _Float16* sh = sh_img[wave];

    // per-lane online softmax state over this lane's 8 outputs/tile
    float run_m = -INFINITY, run_s = 0.f, run_sy = 0.f, run_sx = 0.f;

    // A-fragment addressing: lanes<16: M=lane, K0=0; lanes>=16: M=lane-16, K0=16
    const int r_a = lane & 15;
    const int k0  = hi * 16;

    for (int t = wave; t < NTILES; t += 8) {
        const int y0 = (t / TILES) * 16;
        const int x0 = (t % TILES) * 16;

        // ---- load 32x32 f32 image tile -> f16 LDS (coalesced float4 rows) ----
        {
            const int r0 = lane >> 3;        // 0..3
            const int c4 = (lane & 7) * 4;   // 0,4,...,28
            #pragma unroll
            for (int it = 0; it < 8; ++it) {
                const int r = it * 4 + r0;
                const float4 f =
                    *(const float4*)(img + (size_t)(y0 + r) * WI + x0 + c4);
                union { _Float16 h[4]; unsigned long long u; } pk;
                pk.h[0] = (_Float16)f.x; pk.h[1] = (_Float16)f.y;
                pk.h[2] = (_Float16)f.z; pk.h[3] = (_Float16)f.w;
                *(unsigned long long*)(&sh[r * LDSW + c4]) = pk.u;
            }
        }
        // same-wave LDS RAW: DS pipe is in-order per wave; fence the compiler
        // and drain DScnt explicitly.
        asm volatile("s_wait_dscnt 0x0" ::: "memory");

        // ---- 17 K-slices x {num, s1, s2} WMMAs ----
        v8f cn = {}; v8f c1 = {}; v8f c2 = {};
        #pragma unroll
        for (int dy = 0; dy < HT; ++dy) {
            union { v16h v; uint4 q[2]; } a;
            const _Float16* base = &sh[(dy + r_a) * LDSW + k0];
            a.q[0] = *(const uint4*)(base);        // K k0..k0+7
            a.q[1] = *(const uint4*)(base + 8);    // K k0+8..k0+15
            const v16h av  = a.v;
            const v16h asq = av * av;              // v_pk_mul_f16 x8
            cn = __builtin_amdgcn_wmma_f32_16x16x32_f16(false, av,  false, barr[dy],
                                                        (short)0, cn, false, false);
            c1 = __builtin_amdgcn_wmma_f32_16x16x32_f16(false, av,  false, bones,
                                                        (short)0, c1, false, false);
            c2 = __builtin_amdgcn_wmma_f32_16x16x32_f16(false, asq, false, bones,
                                                        (short)0, c2, false, false);
        }

        // ---- normalize + online softmax update (lane owns 8 (m,n) outputs) ----
        float corr[8];
        float cm = -INFINITY;
        #pragma unroll
        for (int v = 0; v < 8; ++v) {
            const float mu  = c1[v] * INV_N;
            const float var = fmaxf(c2[v] * INV_N - mu * mu, 1e-6f);
            const float cr  = cn[v] * rsqrtf(var);   // num / window_std
            corr[v] = cr;
            cm = fmaxf(cm, cr);
        }
        if (cm > run_m) {
            const float sc = __expf((run_m - cm) * INV_T);
            run_s *= sc; run_sy *= sc; run_sx *= sc;
            run_m = cm;
        }
        #pragma unroll
        for (int v = 0; v < 8; ++v) {
            const float e = __expf((corr[v] - run_m) * INV_T);
            run_s  += e;
            run_sy += e * (float)(y0 + v + 8 * hi);  // D layout: M = v + 8*(lane>>4)
            run_sx += e * (float)(x0 + n);           //           N = lane & 15
        }
    }

    // ---- merge 256 per-thread online-softmax states ----
    sh_red[tid * 4 + 0] = run_m;
    sh_red[tid * 4 + 1] = run_s;
    sh_red[tid * 4 + 2] = run_sy;
    sh_red[tid * 4 + 3] = run_sx;
    __syncthreads();
    if (tid == 0) {
        float M = -INFINITY;
        for (int i = 0; i < 256; ++i) M = fmaxf(M, sh_red[i * 4 + 0]);
        float S = 0.f, SY = 0.f, SX = 0.f;
        for (int i = 0; i < 256; ++i) {
            const float sc = __expf((sh_red[i * 4 + 0] - M) * INV_T);
            S  += sh_red[i * 4 + 1] * sc;
            SY += sh_red[i * 4 + 2] * sc;
            SX += sh_red[i * 4 + 3] * sc;
        }
        const float cc = (float)(LH - 1) * 0.5f;   // 183.5
        out[b * 2 + 0] = SX / S - cc;              // offset_x
        out[b * 2 + 1] = SY / S - cc;              // offset_y
        out[2 * 256 + b] = M;                      // max_corr
    }
}

// ---------------------------------------------------------------------------
extern "C" void kernel_launch(void* const* d_in, const int* in_sizes, int n_in,
                              void* d_out, int out_size, void* d_ws, size_t ws_size,
                              hipStream_t stream) {
    (void)in_sizes; (void)n_in; (void)out_size; (void)ws_size;
    const float* piece = (const float*)d_in[0];   // [256,1,17,17]
    const float* bg    = (const float*)d_in[1];   // [256,1,384,384]
    float* out = (float*)d_out;                   // [256,2] offsets ++ [256] max_corr
    _Float16* tbuf = (_Float16*)d_ws;             // [256][17][48] padded tnorm rows

    tmpl_norm_kernel<<<256, 32, 0, stream>>>(piece, tbuf);
    zncc_kernel<<<256, 256, 0, stream>>>(bg, tbuf, out);
}
